// BackwardEncoder_18056042512499
// MI455X (gfx1250) — matrix operand order repaired and checked by reading, not compiled, not run on hardware
//
#include <hip/hip_runtime.h>
#include <hip/hip_bf16.h>
#include <math.h>

// GRU backward encoder for MI455X (gfx1250), wave32 + bf16 WMMA.
// Phase 1: convert weights to bf16 (ws), zero grid barrier.
// Phase 2: persistent scan kernel, 64 WGs (2 batch halves x 32 col groups),
//          per-WG weight slice cached in 298KB LDS (CDNA5: 320KB/WGP),
//          fuses ig-GEMM + hg-GEMM + gates per step, global barrier per step,
//          prefetches next a-tile, stores hs (bf16) for the output GEMM.
// Phase 3: big WMMA GEMM  out = hs @ w_out^T + b_out.

typedef __bf16 bf16_t;
typedef __attribute__((ext_vector_type(16))) __bf16 v16bf;
typedef __attribute__((ext_vector_type(8)))  float  v8f;

#define B_  32
#define T_  4096
#define P_  512
#define W_  1024
#define G3_ 3072

// LDS row strides (bf16 elems) padded +8 (16B) to avoid bank conflicts
#define LROW_HH 1032
#define LROW_IH 520
#define SCAN_LDS_BYTES ((96 * LROW_HH + 96 * LROW_IH) * 2)   // 297,984 B < 320KB

// workspace layout (bytes)
#define OFF_WIH  0L
#define OFF_WHH  (OFF_WIH  + (long)G3_ * P_ * 2)        // 3,145,728
#define OFF_WOUT (OFF_WHH  + (long)G3_ * W_ * 2)        // 9,437,184
#define OFF_HBF  (OFF_WOUT + (long)P_  * W_ * 2)        // 10,485,760
#define OFF_HF   (OFF_HBF  + 2L * B_ * W_ * 2)          // 10,616,832
#define OFF_BAR  (OFF_HF   + 2L * B_ * W_ * 4)          // 10,878,976
#define OFF_HS   (OFF_BAR  + 256L)                      // 10,879,232  (+256MB hs)

__device__ inline v8f wmma_bf16(v16bf a, v16bf b, v8f c) {
    // D = A(16x32 bf16) * B(32x16 bf16) + C(16x16 f32)
    return __builtin_amdgcn_wmma_f32_16x16x32_bf16(
        false, a, false, b, (short)0, c, false, false);
}

// A fragment (16x32, bf16 source, row-major, row stride in elements).
// lane: m = lane&15, kh = lane>>4 ; VGPR v: kbase = (v>>2)*16 + kh*8 + (v&3)*2
__device__ inline v16bf load_a_bf16(const bf16_t* base, long rowStride, int m, int k0, int kh) {
    const bf16_t* row = base + (long)m * rowStride + k0;
    v16bf a;
#pragma unroll
    for (int v = 0; v < 8; ++v) {
        int kb = ((v >> 2) << 4) + (kh << 3) + ((v & 3) << 1);
        a[2 * v]     = row[kb];
        a[2 * v + 1] = row[kb + 1];
    }
    return a;
}

// Same, but fp32 source converted to bf16 on the fly.
__device__ inline v16bf load_a_f32(const float* base, long rowStride, int m, int k0, int kh) {
    const float* row = base + (long)m * rowStride + k0;
    v16bf a;
#pragma unroll
    for (int v = 0; v < 8; ++v) {
        int kb = ((v >> 2) << 4) + (kh << 3) + ((v & 3) << 1);
        a[2 * v]     = (bf16_t)row[kb];
        a[2 * v + 1] = (bf16_t)row[kb + 1];
    }
    return a;
}

// B fragment (32x16) from a row-major weight [N][K]: B[k][n] = w[nrow][k].
// lane: n = lane&15 (caller folds into nrow), kh = lane>>4 ; VGPR v: kbase = (v>>2)*16 + kh*8 + (v&3)*2
__device__ inline v16bf load_b_w(const bf16_t* w, int rowStride, int nrow, int k0, int kh) {
    const bf16_t* row = w + (long)nrow * rowStride + k0;
    v16bf b;
#pragma unroll
    for (int v = 0; v < 8; ++v) {
        int kb = ((v >> 2) << 4) + (kh << 3) + ((v & 3) << 1);
        b[2 * v]     = row[kb];
        b[2 * v + 1] = row[kb + 1];
    }
    return b;
}

__device__ inline float sigmoidf_(float x) { return 1.0f / (1.0f + expf(-x)); }

__device__ inline void grid_barrier(unsigned* cnt, unsigned gen) {
    __syncthreads();
    if (threadIdx.x == 0) {
        __threadfence();  // make h stores device-visible
        atomicAdd(cnt, 1u);
        unsigned target = gen * gridDim.x;
        while (__hip_atomic_load(cnt, __ATOMIC_ACQUIRE, __HIP_MEMORY_SCOPE_AGENT) < target) {
            __builtin_amdgcn_s_sleep(1);
        }
        __threadfence();
    }
    __syncthreads();
}

// ---------------- Phase 1: weight conversion + barrier init ----------------
__global__ void gru_convert_kernel(const float* w_ih, const float* w_hh, const float* w_out,
                                   bf16_t* w_ih_b, bf16_t* w_hh_b, bf16_t* w_out_b,
                                   unsigned* bar) {
    long i = (long)blockIdx.x * blockDim.x + threadIdx.x;
    if (i == 0) *bar = 0u;
    if (i < (long)G3_ * P_) w_ih_b[i]  = (bf16_t)w_ih[i];
    if (i < (long)G3_ * W_) w_hh_b[i]  = (bf16_t)w_hh[i];
    if (i < (long)P_  * W_) w_out_b[i] = (bf16_t)w_out[i];
}

// ---------------- Phase 2: persistent reverse scan ----------------
// grid = 64 blocks x 64 threads. block = (bh, cg): bh = batch half (16 rows),
// cg = 32-column group of W. Wave owns 16 columns across all 3 gates.
// WG's weight slices live in LDS (copied once): whh [3][32][1024], wih [3][32][512].
__global__ void __launch_bounds__(64)
gru_scan_kernel(const float* __restrict__ a, const float* __restrict__ h0,
                const float* __restrict__ bias, const float* __restrict__ bias_n,
                const bf16_t* __restrict__ w_ih_b, const bf16_t* __restrict__ w_hh_b,
                bf16_t* hbf, float* hf32, bf16_t* hs, unsigned* bar) {
    extern __shared__ bf16_t lds[];
    bf16_t* whh_l = lds;                      // 96 rows x LROW_HH
    bf16_t* wih_l = lds + 96 * LROW_HH;       // 96 rows x LROW_IH

    const int bh   = blockIdx.x & 1;
    const int cg   = blockIdx.x >> 1;          // 0..31
    const int wave = threadIdx.x >> 5;         // 0..1
    const int lane = threadIdx.x & 31;
    const int mA   = lane & 15;   // A-fragment row
    const int kh   = lane >> 4;   // K-half selector
    const int nb   = lane & 15;   // B/D column within tile
    const int colbase = cg * 32 + wave * 16;   // wave's W-column base
    const int wcol    = colbase + nb;          // D-lane's column in W

    // ---- stage this WG's weight slices into LDS (uint4 = 8 bf16) ----
    {
        const uint4* shh = reinterpret_cast<const uint4*>(w_hh_b);
        uint4* dhh = reinterpret_cast<uint4*>(whh_l);
        for (int idx = threadIdx.x; idx < 96 * (W_ / 8); idx += blockDim.x) {
            int row_l = idx >> 7, c = idx & 127;          // 128 uint4 per row
            int g = row_l >> 5, r = row_l & 31;
            long srow = (long)(g * W_ + cg * 32 + r) * (W_ / 8);
            dhh[(long)row_l * (LROW_HH / 8) + c] = shh[srow + c];
        }
        const uint4* sih = reinterpret_cast<const uint4*>(w_ih_b);
        uint4* dih = reinterpret_cast<uint4*>(wih_l);
        for (int idx = threadIdx.x; idx < 96 * (P_ / 8); idx += blockDim.x) {
            int row_l = idx >> 6, c = idx & 63;           // 64 uint4 per row
            int g = row_l >> 5, r = row_l & 31;
            long srow = (long)(g * W_ + cg * 32 + r) * (P_ / 8);
            dih[(long)row_l * (LROW_IH / 8) + c] = sih[srow + c];
        }
    }

    // step-invariant per-lane scalars
    const float br  = bias[0 * W_ + wcol];
    const float bz  = bias[1 * W_ + wcol];
    const float bnI = bias[2 * W_ + wcol];
    const float bnH = bias_n[wcol];

    // init h = broadcast(h0) for this block's region (parity 0)
    for (int idx = threadIdx.x; idx < 16 * 32; idx += blockDim.x) {
        int mrel = idx >> 5, c = idx & 31;
        long o = (long)(bh * 16 + mrel) * W_ + (cg * 32 + c);
        float v = h0[cg * 32 + c];
        hf32[o] = v;
        hbf[o]  = (bf16_t)v;
    }
    __syncthreads();
    grid_barrier(bar, 1u);

    int cur = 0;
    unsigned gen = 2u;
    const long aRowStride = (long)T_ * P_;
    const float* aBatch = a + (long)(bh * 16) * aRowStride;
    const int rowR = 0 * 32 + wave * 16 + nb;   // LDS row for r-gate B frag
    const int rowZ = 1 * 32 + wave * 16 + nb;
    const int rowN = 2 * 32 + wave * 16 + nb;

    for (int t = T_ - 1; t >= 0; --t, ++gen) {
        const int nxt = cur ^ 1;
        v8f accr, accz, accin, acchn;
#pragma unroll
        for (int i = 0; i < 8; ++i) { accr[i] = br; accz[i] = bz; accin[i] = bnI; acchn[i] = bnH; }

        // ---- ig GEMM: a[:,t,:] (16xP) @ W_ih^T slice (B from LDS) ----
        const float* abase = aBatch + (long)t * P_;
#pragma unroll 2
        for (int kt = 0; kt < P_ / 32; ++kt) {
            v16bf af = load_a_f32(abase, aRowStride, mA, kt * 32, kh);
            v16bf b0 = load_b_w(wih_l, LROW_IH, rowR, kt * 32, kh);
            v16bf b1 = load_b_w(wih_l, LROW_IH, rowZ, kt * 32, kh);
            v16bf b2 = load_b_w(wih_l, LROW_IH, rowN, kt * 32, kh);
            accr  = wmma_bf16(af, b0, accr);
            accz  = wmma_bf16(af, b1, accz);
            accin = wmma_bf16(af, b2, accin);
        }

        // prefetch next timestep's a-tile (16 rows x 2KB = 256 cachelines)
        if (t > 0) {
            const float* anext = aBatch + (long)(t - 1) * P_;
            for (int pf = threadIdx.x; pf < 256; pf += blockDim.x) {
                int rrow = pf >> 4, seg = pf & 15;
                __builtin_prefetch(anext + (long)rrow * aRowStride + seg * 32, 0, 0);
            }
        }

        // ---- hg GEMM: h (16xW) @ W_hh^T slice (B from LDS) ----
        const bf16_t* hbase = hbf + (long)cur * B_ * W_ + (long)(bh * 16) * W_;
#pragma unroll 2
        for (int kt = 0; kt < W_ / 32; ++kt) {
            v16bf af = load_a_bf16(hbase, W_, mA, kt * 32, kh);
            v16bf b0 = load_b_w(whh_l, LROW_HH, rowR, kt * 32, kh);
            v16bf b1 = load_b_w(whh_l, LROW_HH, rowZ, kt * 32, kh);
            v16bf b2 = load_b_w(whh_l, LROW_HH, rowN, kt * 32, kh);
            accr  = wmma_bf16(af, b0, accr);
            accz  = wmma_bf16(af, b1, accz);
            acchn = wmma_bf16(af, b2, acchn);
        }

        // ---- gates + state update (lane-local in D layout) ----
        const int mrow0 = (lane >> 4) << 3;
#pragma unroll
        for (int v = 0; v < 8; ++v) {
            int mg = bh * 16 + mrow0 + v;
            float r = sigmoidf_(accr[v]);
            float z = sigmoidf_(accz[v]);
            float n = tanhf(accin[v] + r * acchn[v]);   // accin: ig_n + b_ih_n ; acchn: hg_n + b_n
            long  ho = (long)mg * W_ + wcol;
            float h_old = hf32[(long)cur * B_ * W_ + ho];
            float h_new = n + z * (h_old - n);
            hf32[(long)nxt * B_ * W_ + ho] = h_new;
            hbf [(long)nxt * B_ * W_ + ho] = (bf16_t)h_new;
            hs[((long)mg * T_ + t) * W_ + wcol] = (bf16_t)h_new;
        }

        grid_barrier(bar, gen);
        cur = nxt;
    }
}

// ---------------- Phase 3: out = hs @ w_out^T + b_out ----------------
// 131072 wave-jobs: mt in [0,8192) x nt in [0,16), each wave does 16x32 output.
__global__ void __launch_bounds__(128)
gru_outgemm_kernel(const bf16_t* __restrict__ hs, const bf16_t* __restrict__ w_out_b,
                   const float* __restrict__ b_out, float* __restrict__ out) {
    const int wave = threadIdx.x >> 5;
    const int lane = threadIdx.x & 31;
    const long job = (long)blockIdx.x * 4 + wave;
    const int  mt  = (int)(job >> 4);
    const int  nt  = (int)(job & 15);
    const int  nbase = nt * 32;
    const int  mA = lane & 15, kh = lane >> 4, nb = lane & 15;

    float bo0 = b_out[nbase + nb], bo1 = b_out[nbase + 16 + nb];
    v8f acc0, acc1;
#pragma unroll
    for (int i = 0; i < 8; ++i) { acc0[i] = bo0; acc1[i] = bo1; }

    const bf16_t* abase = hs + (long)mt * 16 * W_;
#pragma unroll 2
    for (int kt = 0; kt < W_ / 32; ++kt) {
        v16bf af = load_a_bf16(abase, W_, mA, kt * 32, kh);
        v16bf b0 = load_b_w(w_out_b, W_, nbase + nb,      kt * 32, kh);
        v16bf b1 = load_b_w(w_out_b, W_, nbase + 16 + nb, kt * 32, kh);
        acc0 = wmma_bf16(af, b0, acc0);
        acc1 = wmma_bf16(af, b1, acc1);
    }

    const int mrow0 = (lane >> 4) << 3;
#pragma unroll
    for (int v = 0; v < 8; ++v) {
        long mg = (long)mt * 16 + mrow0 + v;
        out[mg * P_ + nbase + nb]      = acc0[v];
        out[mg * P_ + nbase + 16 + nb] = acc1[v];
    }
}

extern "C" void kernel_launch(void* const* d_in, const int* in_sizes, int n_in,
                              void* d_out, int out_size, void* d_ws, size_t ws_size,
                              hipStream_t stream) {
    (void)in_sizes; (void)n_in; (void)out_size; (void)ws_size;
    const float* a      = (const float*)d_in[0];
    const float* h0     = (const float*)d_in[1];
    const float* w_ih   = (const float*)d_in[2];
    const float* w_hh   = (const float*)d_in[3];
    const float* bias   = (const float*)d_in[4];
    const float* bias_n = (const float*)d_in[5];
    const float* w_out  = (const float*)d_in[6];
    const float* b_out  = (const float*)d_in[7];
    float* out = (float*)d_out;

    char* ws = (char*)d_ws;
    bf16_t*   w_ih_b  = (bf16_t*)(ws + OFF_WIH);
    bf16_t*   w_hh_b  = (bf16_t*)(ws + OFF_WHH);
    bf16_t*   w_out_b = (bf16_t*)(ws + OFF_WOUT);
    bf16_t*   hbf     = (bf16_t*)(ws + OFF_HBF);
    float*    hf32    = (float*) (ws + OFF_HF);
    unsigned* bar     = (unsigned*)(ws + OFF_BAR);
    bf16_t*   hs      = (bf16_t*)(ws + OFF_HS);

    // allow 298KB dynamic LDS for the scan kernel (gfx1250: up to 320KB/WG)
    (void)hipFuncSetAttribute(reinterpret_cast<const void*>(gru_scan_kernel),
                              hipFuncAttributeMaxDynamicSharedMemorySize,
                              SCAN_LDS_BYTES);

    // Phase 1: convert weights (largest matrix = 3072*1024 elems) + zero barrier
    {
        long n = (long)G3_ * W_;
        int blocks = (int)((n + 255) / 256);
        gru_convert_kernel<<<blocks, 256, 0, stream>>>(w_ih, w_hh, w_out,
                                                       w_ih_b, w_hh_b, w_out_b, bar);
    }
    // Phase 2: persistent reverse GRU scan (64 co-resident workgroups)
    gru_scan_kernel<<<64, 64, SCAN_LDS_BYTES, stream>>>(a, h0, bias, bias_n,
                                                        w_ih_b, w_hh_b,
                                                        hbf, hf32, hs, bar);
    // Phase 3: output projection GEMM
    gru_outgemm_kernel<<<32768, 128, 0, stream>>>(hs, w_out_b, b_out, out);
}
